// Net_43628277792803
// MI455X (gfx1250) — matrix operand order repaired and checked by reading, not compiled, not run on hardware
//
#include <hip/hip_runtime.h>
#include <math.h>

// ---------------------------------------------------------------------------
// GATv2Conv (heads=1) + ELU + log_softmax for MI455X (gfx1250, wave32).
//   N=50000 nodes, E=800000 edges (+N self loops), IN_C=128, OUT_C=64.
// GEMMs: v_wmma_f32_16x16x32_bf16 with pre-packed bf16 operands; per K-step
// the B fragments are batch-loaded (one clause) then 8 WMMAs issue
// back-to-back. Edge phase: L2-resident gathers + float atomics
// (order-invariant segment max via monotone uint keys).
// ---------------------------------------------------------------------------

typedef __attribute__((ext_vector_type(16))) __bf16 v16bf;
typedef __attribute__((ext_vector_type(8)))  __bf16 v8bf;
typedef __attribute__((ext_vector_type(8)))  float  v8f;

#define IN_C  128
#define OUT_C 64
#define NEG_SLOPE 0.2f

// ---- monotone float<->uint key for atomic max over signed floats ----------
__device__ __forceinline__ unsigned fkey(float f) {
    unsigned u = __float_as_uint(f);
    return (u & 0x80000000u) ? ~u : (u | 0x80000000u);
}
__device__ __forceinline__ float funkey(unsigned k) {
    unsigned u = (k & 0x80000000u) ? (k ^ 0x80000000u) : ~k;
    return __uint_as_float(u);
}

// ---------------------------------------------------------------------------
// Pack W_l/W_r (f32 row-major [128][64]) into WMMA B-fragment bf16 layout:
//   frag[mat][kb][t][lane][16],  mat in {L,R}, kb = K/32, t = N/16.
//   lane l -> col n = t*16 + (l&15); elem j -> K = kb*32 + (l<16?0:8) + (j<8 ? j : j+8).
// 1024 threads total; output is 32 KB and stays cache-hot for the GEMM.
// ---------------------------------------------------------------------------
__global__ void pack_weights(const float* __restrict__ Wl,
                             const float* __restrict__ Wr,
                             __bf16* __restrict__ wp)
{
    int tid = blockIdx.x * blockDim.x + threadIdx.x;   // = mat*512 + kb*128 + t*32 + lane
    if (tid >= 1024) return;
    int mat  = tid >> 9;
    int r    = tid & 511;
    int kb   = r >> 7;
    int t    = (r >> 5) & 3;
    int lane = r & 31;

    const float* W = mat ? Wr : Wl;
    int col   = t * 16 + (lane & 15);
    int kbase = kb * 32 + ((lane < 16) ? 0 : 8);

    __bf16* dst = wp + (size_t)tid * 16;
    #pragma unroll
    for (int j = 0; j < 16; ++j)
        dst[j] = (__bf16)W[(size_t)(kbase + (j < 8 ? j : j + 8)) * OUT_C + col];
}

// ---- x (f32) -> row-major bf16; 4 elements/thread (b128 in, b64 out) ------
__global__ void pack_x(const float* __restrict__ x, __bf16* __restrict__ xb,
                       size_t n4)
{
    size_t i = (size_t)blockIdx.x * blockDim.x + threadIdx.x;
    if (i >= n4) return;
    float4 v = ((const float4*)x)[i];
    __bf16* d = xb + i * 4;
    d[0] = (__bf16)v.x; d[1] = (__bf16)v.y;
    d[2] = (__bf16)v.z; d[3] = (__bf16)v.w;
}

// ---------------------------------------------------------------------------
// Kernel: x_l = x@W_l, x_r = x@W_r via WMMA bf16 on packed operands.
// One wave per 16-row tile. Per K-step: 2 b128 A-loads + one clause of
// 16 b128 B-loads (L1-hot) into a fragment array, then 8 WMMAs issued
// back-to-back; 32 v_wmma per wave, 8 f32 accumulators.
// Fragment layouts per CDNA5 ISA 7.12.2:
//   A 16x32 bf16 : lane l -> row m=l&15; elems {0..7} K=base+{0..7},
//                  elems {8..15} K=base+16+{0..7}, base = (l<16?0:8)
//   C/D 16x16 f32: reg r -> row m = r + 8*(l>=16); col n = l&15
// ---------------------------------------------------------------------------
__global__ void gemm_xlr_wmma(const __bf16* __restrict__ xb,
                              const __bf16* __restrict__ wp,
                              float* __restrict__ xl,
                              float* __restrict__ xr, int n)
{
    int wave = (blockIdx.x * blockDim.x + threadIdx.x) >> 5;
    int lane = threadIdx.x & 31;
    int row0 = wave * 16;
    if (row0 >= n) return;

    int m    = lane & 15;
    int half = lane >> 4;             // 0 = lanes 0-15, 1 = lanes 16-31
    int row  = row0 + m;
    if (row >= n) row = n - 1;        // clamp: duplicated rows discarded at store

    const v16bf* wfrag = (const v16bf*)wp;  // index: (mat*16 + kb*4 + t)*32 + lane

    v8f accL[4] = {}, accR[4] = {};

    #pragma unroll
    for (int kb = 0; kb < 4; ++kb) {   // K blocks of 32 (IN_C = 128)
        // ---- A fragment: two aligned 16B loads ----
        const __bf16* xrow = xb + (size_t)row * IN_C + kb * 32 + half * 8;
        v8bf alo = *(const v8bf*)xrow;         // K = base + 0..7
        v8bf ahi = *(const v8bf*)(xrow + 16);  // K = base + 16..23
        v16bf a;
        #pragma unroll
        for (int j = 0; j < 8; ++j) { a[j] = alo[j]; a[j + 8] = ahi[j]; }

        // ---- batch-load all 8 B fragments of this K-step ----
        v16bf bf[8];
        #pragma unroll
        for (int t = 0; t < 4; ++t) {
            bf[t]     = wfrag[(size_t)(( 0 + kb * 4 + t) * 32) + lane];  // W_l
            bf[4 + t] = wfrag[(size_t)((16 + kb * 4 + t) * 32) + lane];  // W_r
        }

        // ---- 8 back-to-back WMMAs ----
        #pragma unroll
        for (int t = 0; t < 4; ++t) {
            accL[t] = __builtin_amdgcn_wmma_f32_16x16x32_bf16(
                          false, a, false, bf[t],     (short)0, accL[t], false, false);
            accR[t] = __builtin_amdgcn_wmma_f32_16x16x32_bf16(
                          false, a, false, bf[4 + t], (short)0, accR[t], false, false);
        }
    }

    // ---- store C/D tiles ----
    #pragma unroll
    for (int t = 0; t < 4; ++t) {
        #pragma unroll
        for (int r = 0; r < 8; ++r) {
            int mrow = row0 + r + half * 8;
            if (mrow < n) {
                size_t o = (size_t)mrow * OUT_C + t * 16 + m;
                xl[o] = accL[t][r];
                xr[o] = accR[t][r];
            }
        }
    }
}

// ---------------------------------------------------------------------------
__global__ void init_node(unsigned* __restrict__ maxkey,
                          float* __restrict__ denom, int n)
{
    int i = blockIdx.x * blockDim.x + threadIdx.x;
    if (i < n) { maxkey[i] = 0u; denom[i] = 0.0f; }   // fkey(-inf) < any real key
}

__global__ void zero_f32(float* __restrict__ p, size_t n)
{
    size_t i = (size_t)blockIdx.x * blockDim.x + threadIdx.x;
    if (i < n) p[i] = 0.0f;
}

// ---------------------------------------------------------------------------
// Per-edge attention logit + atomic segment max over dst.
// One wave per edge: 2 channels/lane, shfl-xor tree reduction (wave32).
// Edges [0,E) from edge_index; [E,E+N) are the implicit self-loops.
// ---------------------------------------------------------------------------
__global__ void edge_logit(const float* __restrict__ xl,
                           const float* __restrict__ xr,
                           const int*   __restrict__ ei,
                           const float* __restrict__ att,
                           float* __restrict__ logit,
                           unsigned* __restrict__ maxkey,
                           int nE, int nN)
{
    int e    = (blockIdx.x * blockDim.x + threadIdx.x) >> 5;
    int lane = threadIdx.x & 31;
    if (e >= nE + nN) return;

    int src, dst;
    if (e < nE) { src = ei[e]; dst = ei[nE + e]; }
    else        { src = dst = e - nE; }

    float v = 0.0f;
    #pragma unroll
    for (int h = 0; h < 2; ++h) {
        int c = lane + 32 * h;
        float t = xl[(size_t)src * OUT_C + c] + xr[(size_t)dst * OUT_C + c];
        t = (t > 0.0f) ? t : NEG_SLOPE * t;    // leaky_relu
        v += t * att[c];
    }
    #pragma unroll
    for (int off = 16; off > 0; off >>= 1) v += __shfl_xor(v, off, 32);

    if (lane == 0) {
        logit[e] = v;
        atomicMax(maxkey + dst, fkey(v));
    }
}

// ---- softmax denominators (one thread per edge) ---------------------------
__global__ void edge_denom(const int* __restrict__ ei,
                           const float* __restrict__ logit,
                           const unsigned* __restrict__ maxkey,
                           float* __restrict__ denom, int nE, int nN)
{
    int e = blockIdx.x * blockDim.x + threadIdx.x;
    if (e >= nE + nN) return;
    int dst = (e < nE) ? ei[nE + e] : (e - nE);
    float mx = funkey(maxkey[dst]);            // every dst has a self loop -> finite
    atomicAdd(denom + dst, expf(logit[e] - mx));
}

// ---- alpha-weighted scatter of source messages (one wave per edge) --------
__global__ void edge_accum(const float* __restrict__ xl,
                           const int*   __restrict__ ei,
                           const float* __restrict__ logit,
                           const unsigned* __restrict__ maxkey,
                           const float* __restrict__ denom,
                           float* __restrict__ acc, int nE, int nN)
{
    int e    = (blockIdx.x * blockDim.x + threadIdx.x) >> 5;
    int lane = threadIdx.x & 31;
    if (e >= nE + nN) return;

    int src, dst;
    if (e < nE) { src = ei[e]; dst = ei[nE + e]; }
    else        { src = dst = e - nE; }

    float mx    = funkey(maxkey[dst]);
    float alpha = expf(logit[e] - mx) / (denom[dst] + 1e-16f);

    atomicAdd(&acc[(size_t)dst * OUT_C + lane],
              xl[(size_t)src * OUT_C + lane] * alpha);
    atomicAdd(&acc[(size_t)dst * OUT_C + lane + 32],
              xl[(size_t)src * OUT_C + lane + 32] * alpha);
}

// ---- bias + ELU + log_softmax over 64 channels (one wave per node) --------
__global__ void node_final(const float* __restrict__ acc,
                           const float* __restrict__ bias,
                           float* __restrict__ out, int nN)
{
    int node = (blockIdx.x * blockDim.x + threadIdx.x) >> 5;
    int lane = threadIdx.x & 31;
    if (node >= nN) return;

    float h0 = acc[(size_t)node * OUT_C + lane]      + bias[lane];
    float h1 = acc[(size_t)node * OUT_C + lane + 32] + bias[lane + 32];
    h0 = (h0 > 0.0f) ? h0 : expm1f(h0);   // ELU
    h1 = (h1 > 0.0f) ? h1 : expm1f(h1);

    float mx = fmaxf(h0, h1);
    #pragma unroll
    for (int off = 16; off > 0; off >>= 1) mx = fmaxf(mx, __shfl_xor(mx, off, 32));
    float s = expf(h0 - mx) + expf(h1 - mx);
    #pragma unroll
    for (int off = 16; off > 0; off >>= 1) s += __shfl_xor(s, off, 32);
    float lg = mx + logf(s);

    out[(size_t)node * OUT_C + lane]      = h0 - lg;
    out[(size_t)node * OUT_C + lane + 32] = h1 - lg;
}

// ---------------------------------------------------------------------------
extern "C" void kernel_launch(void* const* d_in, const int* in_sizes, int n_in,
                              void* d_out, int out_size, void* d_ws, size_t ws_size,
                              hipStream_t stream)
{
    const float* x    = (const float*)d_in[0];
    const int*   ei   = (const int*)  d_in[1];  // edge_index flat (2,E); int32 per harness
    const float* Wl   = (const float*)d_in[2];
    const float* Wr   = (const float*)d_in[3];
    const float* att  = (const float*)d_in[4];
    const float* bias = (const float*)d_in[5];
    float*       out  = (float*)d_out;

    const int N = in_sizes[0] / IN_C;   // 50000
    const int E = in_sizes[1] / 2;      // 800000
    const int T = E + N;                // edges incl. self loops

    // ---- workspace layout (~42.3 MB, all vector-load aligned) ----
    float*    ws     = (float*)d_ws;
    float*    xl     = ws;                           // N*64 f32
    float*    xr     = ws + (size_t)N * OUT_C;       // N*64 f32 (dead after edge_logit)
    float*    logit  = xr + (size_t)N * OUT_C;       // E+N f32
    unsigned* maxkey = (unsigned*)(logit + T);       // N u32
    float*    denom  = (float*)(maxkey + N);         // N f32
    __bf16*   xb     = (__bf16*)(denom + N);         // N*128 bf16 (12.8 MB)
    __bf16*   wp     = xb + (size_t)N * IN_C;        // 2*4*4*32*16 bf16 (32 KB)
    float*    acc    = xr;                           // alias: reuse xr as accumulator

    const int BLK = 256;                             // 8 waves / block

    // 0a) pack weights into WMMA B-fragment bf16 layout (one-shot, 32 KB)
    pack_weights<<<4, BLK, 0, stream>>>(Wl, Wr, wp);
    // 0b) x f32 -> bf16 row-major
    {
        size_t n4 = (size_t)N * IN_C / 4;
        pack_x<<<(int)((n4 + BLK - 1) / BLK), BLK, 0, stream>>>(x, xb, n4);
    }
    // 1) GEMMs on WMMA: one wave per 16-row tile
    {
        int waves = (N + 15) / 16;
        int grid  = (waves * 32 + BLK - 1) / BLK;
        gemm_xlr_wmma<<<grid, BLK, 0, stream>>>(xb, wp, xl, xr, N);
    }
    // 2) per-node init
    init_node<<<(N + BLK - 1) / BLK, BLK, 0, stream>>>(maxkey, denom, N);

    // 3) logits + segment max (one wave / edge)
    {
        int grid = ((T + 7) / 8);   // 8 waves per 256-thread block
        edge_logit<<<grid, BLK, 0, stream>>>(xl, xr, ei, att, logit, maxkey, E, N);
    }
    // 4) zero accumulator (xr is dead now)
    {
        size_t n = (size_t)N * OUT_C;
        zero_f32<<<(int)((n + BLK - 1) / BLK), BLK, 0, stream>>>(acc, n);
    }
    // 5) denominators (one thread / edge)
    edge_denom<<<(T + BLK - 1) / BLK, BLK, 0, stream>>>(ei, logit, maxkey, denom, E, N);

    // 6) weighted scatter (one wave / edge)
    {
        int grid = ((T + 7) / 8);
        edge_accum<<<grid, BLK, 0, stream>>>(xl, ei, logit, maxkey, denom, acc, E, N);
    }
    // 7) ELU + log_softmax (one wave / node)
    {
        int grid = ((N + 7) / 8);
        node_final<<<grid, BLK, 0, stream>>>(acc, bias, out, N);
    }
    (void)n_in; (void)out_size; (void)ws_size;
}
